// SRNN_32057635897703
// MI455X (gfx1250) — matrix operand order repaired and checked by reading, not compiled, not run on hardware
//
#include <hip/hip_runtime.h>
#include <hip/hip_bf16.h>
#include <math.h>

// Problem constants (match reference)
#define T_  128
#define B_  256
#define D_  512
#define H_  1024
#define O_  20
#define OP_ 32                       // O_ padded to WMMA tile multiple
#define THR_   1.0f
#define ALPHA_ 0.9512294245007140f   // exp(-1/20)
#define KAPPA_ 0.9512294245007140f   // exp(-1/20)
#define MROWS_ ((T_ - 1) * B_)
#define NBLK_  33                    // 32 recurrent-GEMM blocks + 1 vo block

typedef __attribute__((ext_vector_type(16))) __bf16 v16bf;
typedef __attribute__((ext_vector_type(8)))  __bf16 v8bf;
typedef __attribute__((ext_vector_type(8)))  float  v8f;
typedef __attribute__((ext_vector_type(4)))  float  v4f;

union V16U { v16bf v; v8bf h[2]; };

static __device__ __forceinline__ v8f wmma_bf16(v16bf a, v16bf b, v8f c) {
  return __builtin_amdgcn_wmma_f32_16x16x32_bf16(false, a, false, b, (short)0, c,
                                                 false, false);
}

// ---------------------------------------------------------------------------
// Prep kernels
// ---------------------------------------------------------------------------
__global__ void split_weights(const float* __restrict__ w,
                              __bf16* __restrict__ hi, __bf16* __restrict__ lo,
                              int n, int diag_dim) {
  int i = blockIdx.x * blockDim.x + threadIdx.x;
  if (i >= n) return;
  float v = w[i];
  if (diag_dim) {
    int r = i / diag_dim, c = i - r * diag_dim;
    if (r == c) v = 0.0f;
  }
  __bf16 h = (__bf16)v;
  hi[i] = h;
  lo[i] = (__bf16)(v - (float)h);
}

// w_out [O_,H_] -> padded [OP_,H_] split hi/lo (rows >= O_ are zero)
__global__ void split_wout_pad(const float* __restrict__ w,
                               __bf16* __restrict__ hi, __bf16* __restrict__ lo) {
  int i = blockIdx.x * blockDim.x + threadIdx.x;
  if (i >= OP_ * H_) return;
  int o = i / H_, hcol = i - o * H_;
  float v = (o < O_) ? w[o * H_ + hcol] : 0.0f;
  __bf16 h = (__bf16)v;
  hi[i] = h;
  lo[i] = (__bf16)(v - (float)h);
}

__global__ void zero_f32(float* p, int n) {
  int i = blockIdx.x * blockDim.x + threadIdx.x;
  if (i < n) p[i] = 0.0f;
}
__global__ void zero_bf16(__bf16* p, int n) {
  int i = blockIdx.x * blockDim.x + threadIdx.x;
  if (i < n) p[i] = (__bf16)0.0f;
}

// ---------------------------------------------------------------------------
// fp32 -> split bf16 A fragment (hi/lo), WMMA 16-bit A layout.
// ---------------------------------------------------------------------------
static __device__ __forceinline__ void load_split_afrag(
    const float* __restrict__ base_row, v16bf& ah, v16bf& al) {
  v4f x0 = *(const v4f*)(base_row + 0);
  v4f x1 = *(const v4f*)(base_row + 4);
  v4f x2 = *(const v4f*)(base_row + 8);
  v4f x3 = *(const v4f*)(base_row + 12);
#pragma unroll
  for (int e = 0; e < 4; ++e) {
    float a0 = x0[e], a1 = x1[e], a2 = x2[e], a3 = x3[e];
    __bf16 h0 = (__bf16)a0, h1 = (__bf16)a1, h2 = (__bf16)a2, h3 = (__bf16)a3;
    ah[e]      = h0; al[e]      = (__bf16)(a0 - (float)h0);
    ah[e + 4]  = h1; al[e + 4]  = (__bf16)(a1 - (float)h1);
    ah[e + 8]  = h2; al[e + 8]  = (__bf16)(a2 - (float)h2);
    ah[e + 12] = h3; al[e + 12] = (__bf16)(a3 - (float)h3);
  }
}

// ---------------------------------------------------------------------------
// Off-critical-path input GEMM over all timesteps:
//   XW[t*B+b, h] = sum_d x[t,b,d] * w_in[h,d]   (split-bf16, 3 passes)
// ---------------------------------------------------------------------------
__global__ __launch_bounds__(256) void xw_gemm(
    const float*  __restrict__ x,
    const __bf16* __restrict__ win_hi, const __bf16* __restrict__ win_lo,
    float* __restrict__ xw) {
  const int lane = threadIdx.x & 31;
  const int wid  = threadIdx.x >> 5;
  const int gw   = blockIdx.x * 8 + wid;
  const int ni   = gw & 63;
  const int mg   = gw >> 6;
  const int m0   = mg * 64;
  const int n0   = ni * 16;
  const int half = lane >> 4;
  const int l15  = lane & 15;
  const int hcol = n0 + l15;

  v8f acc[4] = {};
  for (int k0 = 0; k0 < D_; k0 += 32) {
    v16bf bh = *(const v16bf*)(win_hi + hcol * D_ + k0 + half * 16);
    v16bf bl = *(const v16bf*)(win_lo + hcol * D_ + k0 + half * 16);
    v16bf ah[4], al[4];
#pragma unroll
    for (int s = 0; s < 4; ++s)
      load_split_afrag(x + (size_t)(m0 + s * 16 + l15) * D_ + k0 + half * 16,
                       ah[s], al[s]);
#pragma unroll
    for (int s = 0; s < 4; ++s) {
      acc[s] = wmma_bf16(ah[s], bh, acc[s]);
      acc[s] = wmma_bf16(al[s], bh, acc[s]);
      acc[s] = wmma_bf16(ah[s], bl, acc[s]);
    }
  }
#pragma unroll
  for (int s = 0; s < 4; ++s)
#pragma unroll
    for (int r = 0; r < 8; ++r)
      xw[(size_t)(m0 + s * 16 + r + half * 8) * H_ + hcol] = acc[s][r];
}

// ---------------------------------------------------------------------------
// Device-wide barrier: monotonic counter. Every thread release-fences its own
// stores (drain STOREcnt to device scope) BEFORE block arrival, so thread 0's
// signal implies all of the block's z stores are visible.
// ---------------------------------------------------------------------------
static __device__ __forceinline__ void grid_barrier(int* cnt, int target) {
  __threadfence();                                   // release (all threads)
  __syncthreads();
  if (threadIdx.x == 0) {
    __hip_atomic_fetch_add(cnt, 1, __ATOMIC_ACQ_REL, __HIP_MEMORY_SCOPE_AGENT);
    while (__hip_atomic_load(cnt, __ATOMIC_ACQUIRE, __HIP_MEMORY_SCOPE_AGENT) <
           target)
      __builtin_amdgcn_s_sleep(2);
    __threadfence();                                 // acquire
  }
  __syncthreads();
}

// ---------------------------------------------------------------------------
// Persistent recurrent kernel: ONE launch runs all 128 iterations.
//   Blocks 0..31 (256 waves): v_new = ALPHA*v + XW_t + z@w_rec^T - z*THR
//                             v kept in REGISTERS across all timesteps;
//                             only bf16 spikes z round-trip through memory.
//   Block 32 (8 waves):       vo = KAPPA*vo + z_t @ w_out^T (WMMA, padded N=32)
//                             vo kept in registers; writes out[t].
// ---------------------------------------------------------------------------
__global__ __launch_bounds__(256) void srnn_persistent(
    const float*  __restrict__ xw,       // [(T_-1)*B_, H_]
    const __bf16* __restrict__ wrec_hi,  // [H_, H_] (diag zeroed)
    const __bf16* __restrict__ wrec_lo,
    const __bf16* __restrict__ wout_hi,  // [OP_, H_] padded
    const __bf16* __restrict__ wout_lo,
    __bf16* __restrict__ zb0,            // [B_, H_] double buffer (zb0 zeroed)
    __bf16* __restrict__ zb1,
    float*  __restrict__ out,            // [T_, B_, O_]
    int*    __restrict__ cnt)            // barrier counter (zeroed)
{
  const int lane = threadIdx.x & 31;
  const int wid  = threadIdx.x >> 5;
  const int half = lane >> 4;
  const int l15  = lane & 15;

  if (blockIdx.x < 32) {
    // ------------------ recurrent membrane blocks ------------------
    const int gw = blockIdx.x * 8 + wid;      // 0..255
    const int m0 = (gw & 3) * 64;
    const int n0 = (gw >> 2) * 16;
    const int hcol = n0 + l15;

    v8f vreg[4] = {};                         // membrane state, in registers

    for (int t = 0; t < T_; ++t) {
      if (t < T_ - 1) {
        const __bf16* z_read  = (t & 1) ? zb1 : zb0;
        __bf16*       z_write = (t & 1) ? zb0 : zb1;
        const float*  xw_t    = xw + (size_t)t * B_ * H_;

        // leak + input drive + reset (z_self recomputed from own state)
#pragma unroll
        for (int s = 0; s < 4; ++s)
#pragma unroll
          for (int r = 0; r < 8; ++r) {
            int row = m0 + s * 16 + r + half * 8;
            float vv = vreg[s][r];
            float zs = (vv > THR_) ? 1.0f : 0.0f;
            vreg[s][r] = ALPHA_ * vv - THR_ * zs + xw_t[row * H_ + hcol];
          }

        // recurrent term: z_t @ w_rec_eff^T  (split-bf16, 2 passes)
        for (int k0 = 0; k0 < H_; k0 += 32) {
          v16bf bh = *(const v16bf*)(wrec_hi + hcol * H_ + k0 + half * 16);
          v16bf bl = *(const v16bf*)(wrec_lo + hcol * H_ + k0 + half * 16);
          V16U az[4];
#pragma unroll
          for (int s = 0; s < 4; ++s) {
            const __bf16* pz = z_read + (m0 + s * 16 + l15) * H_ + k0 + half * 16;
            az[s].h[0] = *(const v8bf*)(pz);
            az[s].h[1] = *(const v8bf*)(pz + 8);
          }
#pragma unroll
          for (int s = 0; s < 4; ++s) {
            vreg[s] = wmma_bf16(az[s].v, bh, vreg[s]);
            vreg[s] = wmma_bf16(az[s].v, bl, vreg[s]);
          }
        }

        // publish spikes z_{t+1}
#pragma unroll
        for (int s = 0; s < 4; ++s)
#pragma unroll
          for (int r = 0; r < 8; ++r) {
            int row = m0 + s * 16 + r + half * 8;
            z_write[row * H_ + hcol] =
                (__bf16)((vreg[s][r] > THR_) ? 1.0f : 0.0f);
          }

        // prefetch next timestep's XW tile; overlaps with barrier wait
        if (t + 1 < T_ - 1) {
          const float* xw_n = xw + (size_t)(t + 1) * B_ * H_;
#pragma unroll
          for (int s = 0; s < 4; ++s)
#pragma unroll
            for (int r = 0; r < 8; ++r) {
              int row = m0 + s * 16 + r + half * 8;
              __builtin_prefetch(xw_n + row * H_ + hcol, 0, 3);
            }
        }
      }
      grid_barrier(cnt, NBLK_ * (t + 1));
    }
  } else {
    // ------------------ readout (vo) block: WMMA over padded N=32 ----------
    const int m0 = (wid & 3) * 64;
    const int n0 = (wid >> 2) * 16;          // 0 or 16
    const int ocol = n0 + l15;               // 0..31 (valid if < O_)

    v8f vo[4] = {};                          // vo state, in registers

    for (int t = 0; t < T_; ++t) {
      const __bf16* z_read = (t & 1) ? zb1 : zb0;   // z_t

      // leak
#pragma unroll
      for (int s = 0; s < 4; ++s)
#pragma unroll
        for (int r = 0; r < 8; ++r) vo[s][r] *= KAPPA_;

      // vo += z_t @ w_out^T (split-bf16)
      for (int k0 = 0; k0 < H_; k0 += 32) {
        v16bf bh = *(const v16bf*)(wout_hi + ocol * H_ + k0 + half * 16);
        v16bf bl = *(const v16bf*)(wout_lo + ocol * H_ + k0 + half * 16);
        V16U az[4];
#pragma unroll
        for (int s = 0; s < 4; ++s) {
          const __bf16* pz = z_read + (m0 + s * 16 + l15) * H_ + k0 + half * 16;
          az[s].h[0] = *(const v8bf*)(pz);
          az[s].h[1] = *(const v8bf*)(pz + 8);
        }
#pragma unroll
        for (int s = 0; s < 4; ++s) {
          vo[s] = wmma_bf16(az[s].v, bh, vo[s]);
          vo[s] = wmma_bf16(az[s].v, bl, vo[s]);
        }
      }

      // out[t, b, o] = vo   (only first O_ columns are real)
      if (ocol < O_) {
        float* ot = out + (size_t)t * B_ * O_;
#pragma unroll
        for (int s = 0; s < 4; ++s)
#pragma unroll
          for (int r = 0; r < 8; ++r) {
            int row = m0 + s * 16 + r + half * 8;
            ot[row * O_ + ocol] = vo[s][r];
          }
      }
      grid_barrier(cnt, NBLK_ * (t + 1));
    }
  }
}

// ---------------------------------------------------------------------------
// Fallback path (workspace too small for XW): fused per-step launches.
// ---------------------------------------------------------------------------
__global__ __launch_bounds__(256) void srnn_step_fused(
    const float*  __restrict__ x_t,
    const __bf16* __restrict__ win_hi, const __bf16* __restrict__ win_lo,
    const __bf16* __restrict__ wrec_hi, const __bf16* __restrict__ wrec_lo,
    float* __restrict__ v,
    const __bf16* __restrict__ z_in, __bf16* __restrict__ z_out)
{
  const int lane = threadIdx.x & 31;
  const int wid  = threadIdx.x >> 5;
  const int gw   = blockIdx.x * 8 + wid;
  const int m0   = (gw & 3) * 64;
  const int n0   = (gw >> 2) * 16;
  const int half = lane >> 4;
  const int l15  = lane & 15;
  const int hcol = n0 + l15;

  v8f acc[4];
#pragma unroll
  for (int s = 0; s < 4; ++s)
#pragma unroll
    for (int r = 0; r < 8; ++r) {
      int row = m0 + s * 16 + r + half * 8;
      acc[s][r] = ALPHA_ * v[row * H_ + hcol] - THR_ * (float)z_in[row * H_ + hcol];
    }

  for (int k0 = 0; k0 < D_; k0 += 32) {
    v16bf bh = *(const v16bf*)(win_hi + hcol * D_ + k0 + half * 16);
    v16bf bl = *(const v16bf*)(win_lo + hcol * D_ + k0 + half * 16);
    v16bf ah[4], al[4];
#pragma unroll
    for (int s = 0; s < 4; ++s)
      load_split_afrag(x_t + (size_t)(m0 + s * 16 + l15) * D_ + k0 + half * 16,
                       ah[s], al[s]);
#pragma unroll
    for (int s = 0; s < 4; ++s) {
      acc[s] = wmma_bf16(ah[s], bh, acc[s]);
      acc[s] = wmma_bf16(al[s], bh, acc[s]);
      acc[s] = wmma_bf16(ah[s], bl, acc[s]);
    }
  }

  for (int k0 = 0; k0 < H_; k0 += 32) {
    v16bf bh = *(const v16bf*)(wrec_hi + hcol * H_ + k0 + half * 16);
    v16bf bl = *(const v16bf*)(wrec_lo + hcol * H_ + k0 + half * 16);
    V16U az[4];
#pragma unroll
    for (int s = 0; s < 4; ++s) {
      const __bf16* pz = z_in + (m0 + s * 16 + l15) * H_ + k0 + half * 16;
      az[s].h[0] = *(const v8bf*)(pz);
      az[s].h[1] = *(const v8bf*)(pz + 8);
    }
#pragma unroll
    for (int s = 0; s < 4; ++s) {
      acc[s] = wmma_bf16(az[s].v, bh, acc[s]);
      acc[s] = wmma_bf16(az[s].v, bl, acc[s]);
    }
  }

#pragma unroll
  for (int s = 0; s < 4; ++s)
#pragma unroll
    for (int r = 0; r < 8; ++r) {
      int row = m0 + s * 16 + r + half * 8;
      float vn = acc[s][r];
      v[row * H_ + hcol] = vn;
      z_out[row * H_ + hcol] = (__bf16)((vn > THR_) ? 1.0f : 0.0f);
    }
}

__global__ void vo_step(const __bf16* __restrict__ z,
                        const float*  __restrict__ w_out,
                        float* __restrict__ vo_state, float* __restrict__ out_t) {
  int i = blockIdx.x * blockDim.x + threadIdx.x;
  if (i >= B_ * O_) return;
  int b = i / O_, o = i - b * O_;
  const __bf16* zr = z + b * H_;
  const float*  wr = w_out + o * H_;
  float acc = 0.0f;
  for (int h = 0; h < H_; h += 8) {
    v8bf zc = *(const v8bf*)(zr + h);
    v4f w0 = *(const v4f*)(wr + h);
    v4f w1 = *(const v4f*)(wr + h + 4);
#pragma unroll
    for (int e = 0; e < 4; ++e) acc += (float)zc[e] * w0[e];
#pragma unroll
    for (int e = 0; e < 4; ++e) acc += (float)zc[e + 4] * w1[e];
  }
  float nv = KAPPA_ * vo_state[i] + acc;
  vo_state[i] = nv;
  out_t[i] = nv;
}

// Softmax over last dim (O_=20), in place over [T_*B_, O_]
__global__ void softmax_rows(float* __restrict__ out) {
  int i = blockIdx.x * blockDim.x + threadIdx.x;
  if (i >= T_ * B_) return;
  float* p = out + i * O_;
  float m = p[0];
#pragma unroll
  for (int j = 1; j < O_; ++j) m = fmaxf(m, p[j]);
  float e[O_], s = 0.0f;
#pragma unroll
  for (int j = 0; j < O_; ++j) { e[j] = expf(p[j] - m); s += e[j]; }
  float inv = 1.0f / s;
#pragma unroll
  for (int j = 0; j < O_; ++j) p[j] = e[j] * inv;
}

// ---------------------------------------------------------------------------
extern "C" void kernel_launch(void* const* d_in, const int* in_sizes, int n_in,
                              void* d_out, int out_size, void* d_ws, size_t ws_size,
                              hipStream_t stream) {
  (void)in_sizes; (void)n_in; (void)out_size;
  const float* x     = (const float*)d_in[0];
  const float* w_in  = (const float*)d_in[1];
  const float* w_rec = (const float*)d_in[2];
  const float* w_out = (const float*)d_in[3];
  float* out = (float*)d_out;

  char* ws = (char*)d_ws;
  size_t off = 0;
  __bf16* win_hi  = (__bf16*)(ws + off); off += (size_t)H_ * D_ * 2;
  __bf16* win_lo  = (__bf16*)(ws + off); off += (size_t)H_ * D_ * 2;
  __bf16* wrec_hi = (__bf16*)(ws + off); off += (size_t)H_ * H_ * 2;
  __bf16* wrec_lo = (__bf16*)(ws + off); off += (size_t)H_ * H_ * 2;
  __bf16* wout_hi = (__bf16*)(ws + off); off += (size_t)OP_ * H_ * 2;
  __bf16* wout_lo = (__bf16*)(ws + off); off += (size_t)OP_ * H_ * 2;
  float*  v       = (float*) (ws + off); off += (size_t)B_ * H_ * 4;
  __bf16* z0      = (__bf16*)(ws + off); off += (size_t)B_ * H_ * 2;
  __bf16* z1      = (__bf16*)(ws + off); off += (size_t)B_ * H_ * 2;
  float*  vo      = (float*) (ws + off); off += (size_t)B_ * O_ * 4;
  int*    cnt     = (int*)   (ws + off); off += 256;
  float*  xw      = (float*) (ws + off);
  size_t need_xw  = off + (size_t)MROWS_ * H_ * 4;
  const bool use_xw = (ws_size >= need_xw);   // ws_size fixed -> deterministic

  // --- prep
  split_weights<<<(H_ * D_ + 255) / 256, 256, 0, stream>>>(w_in, win_hi, win_lo,
                                                           H_ * D_, 0);
  split_weights<<<(H_ * H_ + 255) / 256, 256, 0, stream>>>(w_rec, wrec_hi, wrec_lo,
                                                           H_ * H_, H_);
  split_wout_pad<<<(OP_ * H_ + 255) / 256, 256, 0, stream>>>(w_out, wout_hi, wout_lo);
  zero_bf16<<<(B_ * H_ + 255) / 256, 256, 0, stream>>>(z0, B_ * H_);
  zero_f32 <<<1, 1, 0, stream>>>((float*)cnt, 1);

  if (use_xw) {
    // Hoist entire input GEMM off the sequential chain, then run the whole
    // 128-iteration recurrence in ONE persistent kernel (v, vo in registers).
    xw_gemm<<<(MROWS_ / 64) * 64 / 8, 256, 0, stream>>>(x, win_hi, win_lo, xw);
    srnn_persistent<<<NBLK_, 256, 0, stream>>>(xw, wrec_hi, wrec_lo,
                                               wout_hi, wout_lo, z0, z1, out, cnt);
  } else {
    zero_f32<<<(B_ * H_ + 255) / 256, 256, 0, stream>>>(v, B_ * H_);
    zero_f32<<<(B_ * O_ + 255) / 256, 256, 0, stream>>>(vo, B_ * O_);
    zero_f32<<<(B_ * O_ + 255) / 256, 256, 0, stream>>>(out, B_ * O_);
    for (int t = 0; t < T_ - 1; ++t) {
      const __bf16* z_read  = (t & 1) ? z1 : z0;
      __bf16*       z_write = (t & 1) ? z0 : z1;
      srnn_step_fused<<<32, 256, 0, stream>>>(x + (size_t)t * B_ * D_,
                                              win_hi, win_lo, wrec_hi, wrec_lo,
                                              v, z_read, z_write);
      vo_step<<<(B_ * O_ + 255) / 256, 256, 0, stream>>>(
          z_write, w_out, vo, out + (size_t)(t + 1) * B_ * O_);
    }
  }

  softmax_rows<<<(T_ * B_ + 255) / 256, 256, 0, stream>>>(out);
}
